// CLIPVisionEmbeddings_70686571757591
// MI455X (gfx1250) — matrix operand order repaired and checked by reading, not compiled, not run on hardware
//
#include <hip/hip_runtime.h>

typedef __bf16 v4bf  __attribute__((ext_vector_type(4)));
typedef __bf16 v8bf  __attribute__((ext_vector_type(8)));
typedef __bf16 v16bf __attribute__((ext_vector_type(16)));
typedef float  v8f   __attribute__((ext_vector_type(8)));

#define NPATCH 32768
#define PD     768
#define DD     1024
#define BM     64
#define BN     256
#define BK     32
#define LDK    40   // bf16 elems per LDS row (32 data + 8 pad) -> 80B stride, bank-optimal

__device__ __forceinline__ v16bf cat8(v8bf lo, v8bf hi) {
    return __builtin_shufflevector(lo, hi, 0,1,2,3,4,5,6,7,8,9,10,11,12,13,14,15);
}

// split a float4 into bf16 hi + bf16 lo (residual), Ootomo-style
__device__ __forceinline__ void split4(float x0, float x1, float x2, float x3,
                                       v4bf& h, v4bf& l) {
    h[0] = (__bf16)x0; l[0] = (__bf16)(x0 - (float)h[0]);
    h[1] = (__bf16)x1; l[1] = (__bf16)(x1 - (float)h[1]);
    h[2] = (__bf16)x2; l[2] = (__bf16)(x2 - (float)h[2]);
    h[3] = (__bf16)x3; l[3] = (__bf16)(x3 - (float)h[3]);
}

__global__ __launch_bounds__(256) void embed_gemm_kernel(
    const float* __restrict__ patches,   // [N, PD]
    const float* __restrict__ bbox,      // [N, 4]
    const float* __restrict__ g,         // [PD]
    const float* __restrict__ b,         // [PD]
    const float* __restrict__ W,         // [D, PD]
    const float* __restrict__ Wpos,      // [D, 4]
    const float* __restrict__ bpos,      // [D]
    float* __restrict__ out)             // [N+1, D]
{
    __shared__ __attribute__((aligned(16))) __bf16 sAhi[BM * LDK];
    __shared__ __attribute__((aligned(16))) __bf16 sAlo[BM * LDK];
    __shared__ __attribute__((aligned(16))) __bf16 sBhi[BN * LDK];
    __shared__ __attribute__((aligned(16))) __bf16 sBlo[BN * LDK];
    __shared__ float sMu[BM];
    __shared__ float sRs[BM];

    const int tid    = threadIdx.x;
    const int lane   = tid & 31;
    const int wid    = tid >> 5;
    const int l16    = lane & 15;
    const int half   = (lane >> 4) & 1;
    const int wave_m = wid & 3;   // 0..3 -> 16-row strip
    const int wave_n = wid >> 2;  // 0..1 -> 128-col strip

    const int mb   = blockIdx.y;      // 0..511
    const int nb   = blockIdx.x;      // 0..3
    const int grow = mb * BM;
    const int gcol = nb * BN;

    // ---------- Phase 1: LayerNorm stats for 64 rows (4 lanes per row) ----------
    {
        const int row = tid >> 2;   // 0..63
        const int q   = tid & 3;
        const float4* prow = (const float4*)(patches + (size_t)(grow + row) * PD);
        float s = 0.f, ss = 0.f;
        #pragma unroll 4
        for (int i = 0; i < PD / 16; ++i) {      // 48 float4s per lane
            float4 v = prow[q + 4 * i];
            s  += v.x + v.y + v.z + v.w;
            ss += v.x * v.x + v.y * v.y + v.z * v.z + v.w * v.w;
        }
        s  += __shfl_xor(s, 1, 32);  s  += __shfl_xor(s, 2, 32);
        ss += __shfl_xor(ss, 1, 32); ss += __shfl_xor(ss, 2, 32);
        if (q == 0) {
            float mu  = s * (1.0f / PD);
            float var = ss * (1.0f / PD) - mu * mu;
            sMu[row] = mu;
            sRs[row] = rsqrtf(var + 1e-5f);
        }
    }
    __syncthreads();

    v8f acc[8];
    #pragma unroll
    for (int j = 0; j < 8; ++j) acc[j] = (v8f){0.f,0.f,0.f,0.f,0.f,0.f,0.f,0.f};

    // ---------- Phase 2: K loop ----------
    for (int kk = 0; kk < PD; kk += BK) {
        // Stage A (64x32) with LN applied, split into bf16 hi/lo: 2 float4s/thread
        #pragma unroll
        for (int it = 0; it < 2; ++it) {
            int idx = tid + it * 256;
            int row = idx >> 3;
            int c4  = idx & 7;
            int col = kk + c4 * 4;
            float4 v  = *(const float4*)(patches + (size_t)(grow + row) * PD + col);
            float4 gg = *(const float4*)(g + col);
            float4 bb = *(const float4*)(b + col);
            float mu = sMu[row], rs = sRs[row];
            float o0 = (v.x - mu) * rs * gg.x + bb.x;
            float o1 = (v.y - mu) * rs * gg.y + bb.y;
            float o2 = (v.z - mu) * rs * gg.z + bb.z;
            float o3 = (v.w - mu) * rs * gg.w + bb.w;
            v4bf h, l;
            split4(o0, o1, o2, o3, h, l);
            *(v4bf*)(&sAhi[row * LDK + c4 * 4]) = h;
            *(v4bf*)(&sAlo[row * LDK + c4 * 4]) = l;
        }
        // Stage B (256x32 slice of W rows), split into bf16 hi/lo: 8 float4s/thread
        #pragma unroll
        for (int it = 0; it < 8; ++it) {
            int idx = tid + it * 256;
            int d   = idx >> 3;
            int c4  = idx & 7;
            float4 v = *(const float4*)(W + (size_t)(gcol + d) * PD + kk + c4 * 4);
            v4bf h, l;
            split4(v.x, v.y, v.z, v.w, h, l);
            *(v4bf*)(&sBhi[d * LDK + c4 * 4]) = h;
            *(v4bf*)(&sBlo[d * LDK + c4 * 4]) = l;
        }
        __syncthreads();

        // A fragments (16x32 bf16): lane<16 -> K {0..7}U{16..23}; lane>=16 -> K {8..15}U{24..31}
        const int arow = wave_m * 16 + l16;
        v16bf a_hi = cat8(*(const v8bf*)(&sAhi[arow * LDK + half * 8]),
                          *(const v8bf*)(&sAhi[arow * LDK + 16 + half * 8]));
        v16bf a_lo = cat8(*(const v8bf*)(&sAlo[arow * LDK + half * 8]),
                          *(const v8bf*)(&sAlo[arow * LDK + 16 + half * 8]));

        #pragma unroll
        for (int j = 0; j < 8; ++j) {
            // B fragments (32x16 bf16): lane<16 -> K {0..7}U{16..23}; lane>=16 -> K {8..15}U{24..31}
            const int d = wave_n * 128 + j * 16 + l16;
            v16bf b_hi = cat8(*(const v8bf*)(&sBhi[d * LDK + half * 8]),
                              *(const v8bf*)(&sBhi[d * LDK + 16 + half * 8]));
            v16bf b_lo = cat8(*(const v8bf*)(&sBlo[d * LDK + half * 8]),
                              *(const v8bf*)(&sBlo[d * LDK + 16 + half * 8]));
            // bf16x3: hi*hi + hi*lo + lo*hi, f32 accumulate
            acc[j] = __builtin_amdgcn_wmma_f32_16x16x32_bf16(
                false, a_hi, false, b_hi, (short)0, acc[j], false, false);
            acc[j] = __builtin_amdgcn_wmma_f32_16x16x32_bf16(
                false, a_hi, false, b_lo, (short)0, acc[j], false, false);
            acc[j] = __builtin_amdgcn_wmma_f32_16x16x32_bf16(
                false, a_lo, false, b_hi, (short)0, acc[j], false, false);
        }
        __syncthreads();
    }

    // ---------- Phase 3: epilogue — fuse pos-embedding, store ----------
    #pragma unroll
    for (int j = 0; j < 8; ++j) {
        const int col = gcol + wave_n * 128 + j * 16 + l16;
        float4 wp = *(const float4*)(Wpos + (size_t)col * 4);
        float  bp = bpos[col];
        #pragma unroll
        for (int r = 0; r < 8; ++r) {
            const int rowl = wave_m * 16 + half * 8 + r;  // C/D layout: M = r + 8*half
            const int row  = grow + rowl;                 // patch index
            float4 bx = *(const float4*)(bbox + (size_t)row * 4);
            float pos = bx.x * wp.x + bx.y * wp.y + bx.z * wp.z + bx.w * wp.w + bp;
            out[(size_t)(row + 1) * DD + col] = acc[j][r] + pos;
        }
    }
}

__global__ void cls_row_kernel(const float* __restrict__ ce,
                               const float* __restrict__ cp,
                               float* __restrict__ out) {
    int i = blockIdx.x * blockDim.x + threadIdx.x;
    if (i < DD) out[i] = ce[i] + cp[i];
}

extern "C" void kernel_launch(void* const* d_in, const int* in_sizes, int n_in,
                              void* d_out, int out_size, void* d_ws, size_t ws_size,
                              hipStream_t stream) {
    (void)in_sizes; (void)n_in; (void)out_size; (void)d_ws; (void)ws_size;
    const float* patches = (const float*)d_in[0];
    const float* bbox    = (const float*)d_in[1];
    const float* g       = (const float*)d_in[2];
    const float* b       = (const float*)d_in[3];
    const float* W       = (const float*)d_in[4];
    const float* ce      = (const float*)d_in[5];
    const float* Wpos    = (const float*)d_in[6];
    const float* bpos    = (const float*)d_in[7];
    const float* cp      = (const float*)d_in[8];
    float* out = (float*)d_out;

    dim3 grid(DD / BN, NPATCH / BM);  // (4, 512)
    embed_gemm_kernel<<<grid, 256, 0, stream>>>(patches, bbox, g, b, W, Wpos, bpos, out);
    cls_row_kernel<<<(DD + 255) / 256, 256, 0, stream>>>(ce, cp, out);
}
